// KronAttention_34514357190889
// MI455X (gfx1250) — matrix operand order repaired and checked by reading, compiled, not run on hardware
//
#include <hip/hip_runtime.h>
#include <hip/hip_bf16.h>

typedef __bf16 bf16;
typedef __attribute__((ext_vector_type(16))) bf16  v16bf;
typedef __attribute__((ext_vector_type(8)))  bf16  v8bf;
typedef __attribute__((ext_vector_type(8)))  float v8f;
typedef __attribute__((ext_vector_type(4)))  unsigned int u32x4;
typedef __attribute__((ext_vector_type(4)))  int i32x4;
typedef __attribute__((ext_vector_type(8)))  int i32x8;

#define WMMA_BF16(A,B,C) __builtin_amdgcn_wmma_f32_16x16x32_bf16(false,(A),false,(B),(short)0,(C),false,false)
#define SHUF16(lo,hi) __builtin_shufflevector((lo),(hi),0,1,2,3,4,5,6,7,8,9,10,11,12,13,14,15)

// TDM builtin: 5-arg on ROCm 7.2 (clang-22), 6-arg on amdgpu-toolchain (clang-23).
#if defined(__HIP_DEVICE_COMPILE__) && __has_builtin(__builtin_amdgcn_tensor_load_to_lds)
  #define USE_TDM 1
  #if (__clang_major__ >= 23)
    #define TDM_6ARG 1
  #else
    #define TDM_6ARG 0
  #endif
#else
  #define USE_TDM 0
#endif

static constexpr int Bsz = 4, Sq = 2048, Dm = 1024, Hn = 16, HD = 64;
static constexpr int Mrows = Bsz * Sq;   // 8192

// ---------------- conversion kernels ----------------
__global__ __launch_bounds__(256)
void cvt_f32_bf16(const float* __restrict__ x, bf16* __restrict__ y, int n) {
    int i = blockIdx.x * 256 + threadIdx.x;
    if (i < n) y[i] = (bf16)x[i];
}

// W[K,N] fp32 -> Wt[N,K] bf16  (coalesced read along N)
__global__ __launch_bounds__(256)
void cvt_w_transpose(const float* __restrict__ w, bf16* __restrict__ wt) {
    int i = blockIdx.x * 256 + threadIdx.x;        // 0 .. 1M-1
    int k = i >> 10, n = i & 1023;
    wt[(size_t)n * Dm + k] = (bf16)w[i];           // w[i] == w[k*1024+n]
}

// ---------------- TDM helpers -----------------------
#if USE_TDM
// Pack a 2-D tile descriptor (D# groups 0/1) per CDNA5 ISA
// and issue TENSOR_LOAD_TO_LDS. data_size = 2 bytes. LDS row padding enabled.
__device__ __forceinline__ void tdm_load_2d(const void* gaddr, unsigned ldsaddr,
                                            unsigned tens_d0, unsigned tens_d1,
                                            unsigned stride0,
                                            unsigned tile_d0, unsigned tile_d1,
                                            unsigned pad_interval, unsigned pad_amount)
{
    unsigned long long ga = (unsigned long long)(size_t)gaddr;
    u32x4 g0;
    g0[0] = 1u;                                        // count=1 (valid), user mode
    g0[1] = ldsaddr;                                   // lds_addr [63:32]
    g0[2] = (unsigned)(ga & 0xffffffffu);              // global_addr [95:64]
    g0[3] = (unsigned)((ga >> 32) & 0x01ffffffu)       // global_addr [120:96]
          | (2u << 30);                                // type=2 ("image")
    i32x8 g1;
    unsigned d0 = (1u << 16)                           // data_size=1 -> 2 bytes
                | (1u << 20)                           // pad_enable
                | (pad_interval << 22) | (pad_amount << 25);
    g1[0] = (int)d0;
    g1[1] = (int)((tens_d0 & 0xffffu) << 16);          // tensor_dim0[15:0]  @63:48
    g1[2] = (int)(((tens_d0 >> 16) & 0xffffu)          // tensor_dim0[31:16] @79:64
          | ((tens_d1 & 0xffffu) << 16));              // tensor_dim1[15:0]  @95:80
    g1[3] = (int)(((tens_d1 >> 16) & 0xffffu)          // tensor_dim1[31:16] @111:96
          | ((tile_d0 & 0xffffu) << 16));              // tile_dim0          @127:112
    g1[4] = (int)(tile_d1 & 0xffffu);                  // tile_dim1 (tile_dim2=0)
    g1[5] = (int)stride0;                              // tensor_dim0_stride[31:0]
    g1[6] = 0;                                         // stride0 hi / stride1 lo
    g1[7] = 0;
    i32x4 z4 = {0, 0, 0, 0};
#if TDM_6ARG
    i32x8 z8 = {0, 0, 0, 0, 0, 0, 0, 0};
    __builtin_amdgcn_tensor_load_to_lds(g0, g1, z4, z4, z8, 0);
#else
    __builtin_amdgcn_tensor_load_to_lds(g0, g1, z4, z4, 0);
#endif
}
__device__ __forceinline__ unsigned lds_addr_of(const void* p) {
    return (unsigned)(size_t)p;   // flat LDS address: low 32 bits = LDS offset
}
__device__ __forceinline__ void tdm_wait_all() {
#if __has_builtin(__builtin_amdgcn_s_wait_tensorcnt)
    __builtin_amdgcn_s_wait_tensorcnt(0);
#else
    asm volatile("s_wait_tensorcnt 0x0" ::: "memory");
#endif
}
#endif

// ---------------- projection GEMM -------------------
// 32x64 tile per wave: 8 accumulators, each B-operand load feeds 2 WMMAs.
//  mode 0: Q -> [B,H,S,HD] bf16, scaled      mode 1: K -> [B,H,S,HD] bf16
//  mode 2: V -> [B,H,HD,S] bf16 (transposed) mode 3: out -> [M,N] fp32
__global__ __launch_bounds__(128)
void gemm_qkvo(const bf16* __restrict__ A, const bf16* __restrict__ Wt,
               const float* __restrict__ bias, bf16* __restrict__ outb,
               float* __restrict__ outf, int mode, float scale)
{
    const int lane = threadIdx.x & 31;
    const int wid  = threadIdx.x >> 5;
    const int gw   = blockIdx.x * 4 + wid;
    const int ntn  = Dm / 64;                      // 16 N-tiles
    const int row0 = (gw / ntn) * 32;
    const int n0   = (gw % ntn) * 64;
    const int m_ = lane & 15, h_ = lane >> 4, n_ = m_;

    v8f acc[2][4] = {};
    const bf16* arow0 = A + (size_t)(row0 + m_) * Dm;
    const bf16* arow1 = arow0 + (size_t)16 * Dm;

    for (int k0 = 0; k0 < Dm; k0 += 32) {
        __builtin_prefetch(arow0 + k0 + 512, 0, 3);
        __builtin_prefetch(arow1 + k0 + 512, 0, 3);
        v16bf av0 = SHUF16(*(const v8bf*)(arow0 + k0 + 16 * h_),
                           *(const v8bf*)(arow0 + k0 + 16 * h_ + 8));
        v16bf av1 = SHUF16(*(const v8bf*)(arow1 + k0 + 16 * h_),
                           *(const v8bf*)(arow1 + k0 + 16 * h_ + 8));
        const bf16* wbase = Wt + (size_t)n0 * Dm + k0 + 16 * h_;
        #pragma unroll
        for (int j = 0; j < 4; ++j) {
            v16bf bv = *(const v16bf*)(wbase + (size_t)(j * 16 + n_) * Dm);
            acc[0][j] = WMMA_BF16(av0, bv, acc[0][j]);
            acc[1][j] = WMMA_BF16(av1, bv, acc[1][j]);
        }
    }

    #pragma unroll
    for (int i = 0; i < 2; ++i) {
        #pragma unroll
        for (int j = 0; j < 4; ++j) {
            #pragma unroll
            for (int r = 0; r < 8; ++r) {
                int row = row0 + i * 16 + r + 8 * h_;
                int col = n0 + j * 16 + n_;
                float v = (acc[i][j][r] + bias[col]) * scale;
                if (mode == 3) {
                    outf[(size_t)row * Dm + col] = v;
                } else {
                    int b = row >> 11, s = row & 2047;           // S = 2048
                    int head = col >> 6, hd = col & 63;          // HD = 64
                    size_t bh = (size_t)(b * Hn + head);
                    if (mode == 2) outb[(bh * HD + hd) * Sq + s] = (bf16)v;
                    else           outb[(bh * Sq + s) * HD + hd] = (bf16)v;
                }
            }
        }
    }
}

// ---------------- flash attention -------------------
// one wave = one (b,h) 16-query tile; keys in tiles of 32; K/V tiles staged
// into LDS by the Tensor Data Mover (double-buffered), P transposed via LDS.
__global__ __launch_bounds__(64)
void flash16(const bf16* __restrict__ Qs, const bf16* __restrict__ Ks,
             const bf16* __restrict__ Vt, bf16* __restrict__ Ao)
{
    constexpr int KSTR = 72;   // 64 + 8 pad bf16 -> 36-dword rows, bank-conflict free
    constexpr int VSTR = 40;   // 32 + 8 pad bf16 -> 20-dword rows, bank-conflict free
    struct WaveSmem {
        bf16 kt[2][32 * KSTR];
        bf16 vt[2][64 * VSTR];
        bf16 p[16 * 40];
    };
    __shared__ WaveSmem smem[2];

    const int lane = threadIdx.x & 31;
    const int wid  = threadIdx.x >> 5;
    const int gw   = blockIdx.x * 2 + wid;
    const int bh   = gw >> 7;                      // S/16 = 128 q-tiles per (b,h)
    const int q0   = (gw & 127) * 16;
    const int b    = bh >> 4, head = bh & 15;
    const int m_ = lane & 15, h_ = lane >> 4, n_ = m_;

    const bf16* Qb = Qs + (size_t)bh * Sq * HD;
    const bf16* Kb = Ks + (size_t)bh * Sq * HD;
    const bf16* Vb = Vt + (size_t)bh * HD * Sq;
    WaveSmem& sm = smem[wid];
    bf16* pb = sm.p;

    v16bf qa[2];
    {
        const bf16* qrow = Qb + (size_t)(q0 + m_) * HD;
        #pragma unroll
        for (int c = 0; c < 2; ++c)
            qa[c] = SHUF16(*(const v8bf*)(qrow + c * 32 + 16 * h_),
                           *(const v8bf*)(qrow + c * 32 + 16 * h_ + 8));
    }

    float rm[8], rl[8];
    #pragma unroll
    for (int r = 0; r < 8; ++r) { rm[r] = -3.0e38f; rl[r] = 0.0f; }
    v8f o[4] = {v8f{}, v8f{}, v8f{}, v8f{}};

    const int nk = (q0 + 16 + 31) >> 5;            // number of 32-key tiles

#if USE_TDM
    auto issue_tile = [&](int t) {
        const int k0 = t * 32, bi = t & 1;
        // K tile: 32 rows x 64 feats, row = 32 dwords -> pad_interval code 4, +4 dwords
        tdm_load_2d(Kb + (size_t)k0 * HD, lds_addr_of(sm.kt[bi]),
                    HD, (unsigned)(Sq - k0), HD, HD, 32, 4, 3);
        // V tile: 64 rows x 32 keys, row = 16 dwords -> pad_interval code 3, +4 dwords
        tdm_load_2d(Vb + k0, lds_addr_of(sm.vt[bi]),
                    (unsigned)(Sq - k0), HD, Sq, 32, HD, 3, 3);
    };
    issue_tile(0);
#endif

    for (int t = 0; t < nk; ++t) {
        const int k0 = t * 32;
#if USE_TDM
        tdm_wait_all();                            // tile t resident in LDS
        if (t + 1 < nk) issue_tile(t + 1);         // overlap DMA with compute
        const bf16* kbase = sm.kt[t & 1]; const int kstr = KSTR;
        const bf16* vbase = sm.vt[t & 1]; const int vstr = VSTR;
#else
        const bf16* kbase = Kb + (size_t)k0 * HD;  const int kstr = HD;
        const bf16* vbase = Vb + k0;               const int vstr = Sq;
#endif
        // scores: S[16x32] = Q(16x64) @ K^T
        v8f sc[2];
        #pragma unroll
        for (int s2 = 0; s2 < 2; ++s2) {
            const bf16* krow = kbase + (size_t)(s2 * 16 + n_) * kstr + 16 * h_;
            v16bf kb0 = *(const v16bf*)(krow);
            v16bf kb1 = *(const v16bf*)(krow + 32);
            v8f z = {};
            sc[s2] = WMMA_BF16(qa[0], kb0, z);
            sc[s2] = WMMA_BF16(qa[1], kb1, sc[s2]);
        }
        // causal mask (only partially-masked tiles)
        if (k0 + 31 > q0) {
            #pragma unroll
            for (int s2 = 0; s2 < 2; ++s2)
                #pragma unroll
                for (int r = 0; r < 8; ++r) {
                    int ki = k0 + s2 * 16 + n_;
                    int qi = q0 + r + 8 * h_;
                    if (ki > qi) sc[s2][r] = -3.0e38f;
                }
        }
        // online softmax (row stats replicated across each 16-lane half)
        float alpha[8];
        #pragma unroll
        for (int r = 0; r < 8; ++r) {
            float tmx = fmaxf(sc[0][r], sc[1][r]);
            #pragma unroll
            for (int mk = 1; mk < 16; mk <<= 1) tmx = fmaxf(tmx, __shfl_xor(tmx, mk, 32));
            float mn = fmaxf(rm[r], tmx);
            alpha[r] = __expf(rm[r] - mn);
            rm[r] = mn;
            float p0 = __expf(sc[0][r] - mn);
            float p1 = __expf(sc[1][r] - mn);
            sc[0][r] = p0; sc[1][r] = p1;
            float ps = p0 + p1;
            #pragma unroll
            for (int mk = 1; mk < 16; mk <<= 1) ps += __shfl_xor(ps, mk, 32);
            rl[r] = rl[r] * alpha[r] + ps;
        }
        #pragma unroll
        for (int j = 0; j < 4; ++j)
            #pragma unroll
            for (int r = 0; r < 8; ++r) o[j][r] *= alpha[r];

        // C-layout -> A-layout transpose through LDS (in-order per wave)
        #pragma unroll
        for (int s2 = 0; s2 < 2; ++s2)
            #pragma unroll
            for (int r = 0; r < 8; ++r)
                pb[(r + 8 * h_) * 40 + s2 * 16 + n_] = (bf16)sc[s2][r];
        v16bf pa;
        {
            const bf16* prow = pb + m_ * 40;
            pa = SHUF16(*(const v8bf*)(prow + 16 * h_),
                        *(const v8bf*)(prow + 16 * h_ + 8));
        }
        // O += P @ V  (V rows = features, keys contiguous)
        #pragma unroll
        for (int j = 0; j < 4; ++j) {
            v16bf vb = *(const v16bf*)(vbase + (size_t)(j * 16 + n_) * vstr + 16 * h_);
            o[j] = WMMA_BF16(pa, vb, o[j]);
        }
    }

    // epilogue: normalize, merge heads -> [B,S,D] bf16
    float inv[8];
    #pragma unroll
    for (int r = 0; r < 8; ++r) inv[r] = 1.0f / rl[r];
    #pragma unroll
    for (int j = 0; j < 4; ++j)
        #pragma unroll
        for (int r = 0; r < 8; ++r) {
            int row = q0 + r + 8 * h_;
            int col = head * 64 + j * 16 + n_;
            Ao[((size_t)(b * Sq + row)) * Dm + col] = (bf16)(o[j][r] * inv[r]);
        }
}

// ---------------- launcher --------------------------
extern "C" void kernel_launch(void* const* d_in, const int* in_sizes, int n_in,
                              void* d_out, int out_size, void* d_ws, size_t ws_size,
                              hipStream_t stream)
{
    const float* hs = (const float*)d_in[0];
    const float* Wq = (const float*)d_in[1];
    const float* bq = (const float*)d_in[2];
    const float* Wk = (const float*)d_in[3];
    const float* bk = (const float*)d_in[4];
    const float* Wv = (const float*)d_in[5];
    const float* bv = (const float*)d_in[6];
    const float* Wo = (const float*)d_in[7];
    const float* bo = (const float*)d_in[8];
    float* out = (float*)d_out;

    char* ws = (char*)d_ws;
    size_t off = 0;
    auto alloc = [&](size_t bytes) -> void* {
        void* p = ws + off;
        off += (bytes + 255) & ~(size_t)255;
        return p;
    };
    bf16* Xb   = (bf16*)alloc((size_t)Mrows * Dm * 2);   // 16 MB
    bf16* Wqt  = (bf16*)alloc((size_t)Dm * Dm * 2);      //  2 MB
    bf16* Wkt  = (bf16*)alloc((size_t)Dm * Dm * 2);
    bf16* Wvt  = (bf16*)alloc((size_t)Dm * Dm * 2);
    bf16* Wot  = (bf16*)alloc((size_t)Dm * Dm * 2);
    bf16* Qb   = (bf16*)alloc((size_t)Mrows * Dm * 2);   // [B,H,S,HD]
    bf16* Kb   = (bf16*)alloc((size_t)Mrows * Dm * 2);   // [B,H,S,HD]
    bf16* Vtb  = (bf16*)alloc((size_t)Mrows * Dm * 2);   // [B,H,HD,S]
    bf16* Attn = (bf16*)alloc((size_t)Mrows * Dm * 2);   // [B,S,D]

    const int nElem = Mrows * Dm;                        // 8,388,608
    cvt_f32_bf16<<<nElem / 256, 256, 0, stream>>>(hs, Xb, nElem);
    cvt_w_transpose<<<(Dm * Dm) / 256, 256, 0, stream>>>(Wq, Wqt);
    cvt_w_transpose<<<(Dm * Dm) / 256, 256, 0, stream>>>(Wk, Wkt);
    cvt_w_transpose<<<(Dm * Dm) / 256, 256, 0, stream>>>(Wv, Wvt);
    cvt_w_transpose<<<(Dm * Dm) / 256, 256, 0, stream>>>(Wo, Wot);

    // (M/32)*(N/64) = 4096 waves / 4 per block = 1024 blocks
    gemm_qkvo<<<1024, 128, 0, stream>>>(Xb, Wqt, bq, Qb, nullptr, 0, 0.125f); // 1/sqrt(64)
    gemm_qkvo<<<1024, 128, 0, stream>>>(Xb, Wkt, bk, Kb, nullptr, 1, 1.0f);
    gemm_qkvo<<<1024, 128, 0, stream>>>(Xb, Wvt, bv, Vtb, nullptr, 2, 1.0f);

    // B*H*(S/16) = 8192 waves / 2 per block = 4096 blocks
    flash16<<<4096, 64, 0, stream>>>(Qb, Kb, Vtb, Attn);

    gemm_qkvo<<<1024, 128, 0, stream>>>(Attn, Wot, bo, nullptr, out, 3, 1.0f);
}
